// Attention_52012053955205
// MI455X (gfx1250) — matrix operand order repaired and checked by reading, not compiled, not run on hardware
//
#include <hip/hip_runtime.h>
#include <hip/hip_bf16.h>
#include <cstdint>

// ---------------------------------------------------------------------------
// Problem constants (from reference): B=2, S=2048, D=1024, H=16, HD=64
// ---------------------------------------------------------------------------
constexpr int Bc  = 2;
constexpr int Sc  = 2048;
constexpr int Dc  = 1024;
constexpr int Hc  = 16;
constexpr int HDc = 64;
constexpr int ROWS = Bc * Sc;      // 4096 (B*S rows)
constexpr int HHD  = Hc * HDc;     // 1024 (H*HD columns)

typedef __attribute__((ext_vector_type(16))) __bf16        v16bf;
typedef __attribute__((ext_vector_type(2)))  __bf16        v2bf;
typedef __attribute__((ext_vector_type(8)))  float         v8f;
typedef __attribute__((ext_vector_type(4)))  float         v4f;
typedef __attribute__((ext_vector_type(4)))  unsigned int  v4u;
typedef __attribute__((ext_vector_type(2)))  unsigned int  v2u;

union Frag16 {
    unsigned short u[16];
    v4u            q[2];     // two 16-byte halves -> ds_load_b128 pairs
    v16bf          v;
};

union Pack8 { v4u q; unsigned short us[8]; };

// Packed f32x2 -> bf16x2 conversion; single V_CVT_PK_BF16_F32 when available.
static __device__ __forceinline__ unsigned int cvt2bf(float lo, float hi) {
#if __has_builtin(__builtin_amdgcn_cvt_pk_bf16_f32)
    union { v2bf v; unsigned int u; } c;
    c.v = __builtin_amdgcn_cvt_pk_bf16_f32(lo, hi);
    return c.u;
#else
    union { float f; unsigned int u; } a, b; a.f = lo; b.f = hi;
    unsigned int ra = a.u + 0x7FFFu + ((a.u >> 16) & 1u);
    unsigned int rb = b.u + 0x7FFFu + ((b.u >> 16) & 1u);
    return (ra >> 16) | (rb & 0xFFFF0000u);
#endif
}

static __device__ __forceinline__ unsigned short f2bf(float f) {
    return (unsigned short)(cvt2bf(f, f) & 0xFFFFu);
}

static __device__ __forceinline__ v8f vzero8() {
    v8f z;
    #pragma unroll
    for (int i = 0; i < 8; ++i) z[i] = 0.0f;
    return z;
}

// Low 32 bits of a flat shared-memory pointer = offset in the group segment;
// DS ops and async-to-LDS both add the same per-wave LDS_BASE.
static __device__ __forceinline__ unsigned lds_off32(const void* p) {
    return (unsigned)(size_t)p;
}

// Async 16-byte copy global -> LDS (gfx1250): tracked on ASYNCcnt, no VGPR data.
static __device__ __forceinline__ void async_copy_b128(unsigned lds_addr,
                                                       const void* gaddr) {
    asm volatile("global_load_async_to_lds_b128 %0, %1, off"
                 :: "v"(lds_addr), "v"(gaddr) : "memory");
}

static __device__ __forceinline__ void wait_async0() {
    asm volatile("s_wait_asynccnt 0" ::: "memory");
}

// LDS row stride (ushorts) for K=64 tiles: 144 B rows (16B-aligned, 36 banks,
// period-16 mod 64 -> conflict-free across a 16-lane half).
constexpr int LDK = 72;

// ---------------------------------------------------------------------------
// Kernel 1: fused QKV projection.  out[bs, h*64+d] = x @ W[h] + bias[h]  (bf16)
// Block = 256 threads (8 waves); 128x64 output tile; K-loop step 64.
// ---------------------------------------------------------------------------
__global__ __launch_bounds__(256) void qkv_proj_kernel(
    const float* __restrict__ x,
    const float* __restrict__ Wq, const float* __restrict__ Wk, const float* __restrict__ Wv,
    const float* __restrict__ bq, const float* __restrict__ bk, const float* __restrict__ bv,
    unsigned short* __restrict__ qws, unsigned short* __restrict__ kws,
    unsigned short* __restrict__ vws)
{
    __shared__ __align__(16) unsigned short As [128][LDK];  // [M][K]
    __shared__ __align__(16) unsigned short BsT[64][LDK];   // [N][K] (transposed)

    const int tid  = threadIdx.x;
    const int wave = tid >> 5;
    const int lane = tid & 31;
    const int m0   = blockIdx.x * 128;
    const int h    = blockIdx.y;
    const int w    = blockIdx.z;

    const float* Wm   = (w == 0) ? Wq : (w == 1) ? Wk : Wv;
    const float* bias = (((w == 0) ? bq : (w == 1) ? bk : bv)) + h * HDc;
    unsigned short* out = (w == 0) ? qws : (w == 1) ? kws : vws;
    Wm += (size_t)h * Dc * HDc;

    v8f acc[4];
    #pragma unroll
    for (int nt = 0; nt < 4; ++nt) acc[nt] = vzero8();

    const int nl  = lane & 15;
    const int am  = wave * 16 + nl;          // A-matrix M row for this lane
    const int kb  = (lane & 16) ? 8 : 0;     // A-fragment K base (ISA 7.12.2)
    const int bko = (lane & 16) ? 8 : 0;     // B-fragment K base

    for (int k0 = 0; k0 < Dc; k0 += 64) {
        // stage A: 128x64 fp32 -> bf16 (float4 in, b64 out); exact trip count
        #pragma unroll
        for (int it = 0; it < 8; ++it) {
            int i = tid + it * 256;
            int r = i >> 4, c4 = (i & 15) * 4;
            v4f xv = *(const v4f*)&x[(size_t)(m0 + r) * Dc + (k0 + c4)];
            v2u p; p[0] = cvt2bf(xv[0], xv[1]); p[1] = cvt2bf(xv[2], xv[3]);
            *(v2u*)&As[r][c4] = p;
        }
        // stage B transposed: W[k][n] -> BsT[n][k]
        #pragma unroll
        for (int it = 0; it < 4; ++it) {
            int i = tid + it * 256;
            int r = i >> 4, c4 = (i & 15) * 4;
            v4f wv = *(const v4f*)&Wm[(size_t)(k0 + r) * HDc + c4];
            #pragma unroll
            for (int jx = 0; jx < 4; ++jx) BsT[c4 + jx][r] = f2bf(wv[jx]);
        }
        __syncthreads();

        #pragma unroll
        for (int ks = 0; ks < 2; ++ks) {
            Frag16 a;
            a.q[0] = *(const v4u*)&As[am][ks * 32 + kb];
            a.q[1] = *(const v4u*)&As[am][ks * 32 + kb + 16];
            #pragma unroll
            for (int nt = 0; nt < 4; ++nt) {
                Frag16 bfr;
                bfr.q[0] = *(const v4u*)&BsT[nt * 16 + nl][ks * 32 + bko];
                bfr.q[1] = *(const v4u*)&BsT[nt * 16 + nl][ks * 32 + bko + 16];
                acc[nt] = __builtin_amdgcn_wmma_f32_16x16x32_bf16(
                    false, a.v, false, bfr.v, (short)0, acc[nt], false, false);
            }
        }
        __syncthreads();
    }

    const int rbase = m0 + wave * 16 + ((lane & 16) ? 8 : 0);
    #pragma unroll
    for (int nt = 0; nt < 4; ++nt) {
        int col = nt * 16 + nl;
        float bb = bias[col];
        #pragma unroll
        for (int r = 0; r < 8; ++r)
            out[(size_t)(rbase + r) * HHD + h * HDc + col] = f2bf(acc[nt][r] + bb);
    }
}

// ---------------------------------------------------------------------------
// Kernel 2: causal flash attention per (batch, head, 128-row q tile).
// 64-wide k-seq tiles; Q/K staged via async-to-LDS (ASYNCcnt), V transposed
// manually (async can't transpose).
// ---------------------------------------------------------------------------
__global__ __launch_bounds__(256) void attn_kernel(
    const unsigned short* __restrict__ qws,
    const unsigned short* __restrict__ kws,
    const unsigned short* __restrict__ vws,
    unsigned short* __restrict__ zws)
{
    __shared__ __align__(16) unsigned short Qs [128][LDK];   // [M][K=hd]
    __shared__ __align__(16) unsigned short Ks [64][LDK];    // [N=kseq][K=hd]
    __shared__ __align__(16) unsigned short VsT[64][LDK];    // [N=hd][K=kseq]
    __shared__ __align__(16) unsigned short Ps [8][16][LDK]; // per-wave [M][K=kseq]

    const int tid  = threadIdx.x;
    const int wave = tid >> 5;
    const int lane = tid & 31;
    const int q0   = blockIdx.x * 128;
    const int h    = blockIdx.y;
    const int b    = blockIdx.z;

    const size_t base = ((size_t)b * Sc) * HHD + (size_t)h * HDc;

    const unsigned qs_base = lds_off32(&Qs[0][0]);
    const unsigned ks_base = lds_off32(&Ks[0][0]);

    // stage Q: async global->LDS b128 copies; exact trip count
    #pragma unroll
    for (int it = 0; it < 4; ++it) {
        int i = tid + it * 256;
        int r = i >> 3, c8 = (i & 7) * 8;
        async_copy_b128(qs_base + (unsigned)(r * LDK + c8) * 2u,
                        &qws[base + (size_t)(q0 + r) * HHD + c8]);
    }
    wait_async0();
    __syncthreads();

    const int nl  = lane & 15;
    const int hi8 = (lane & 16) ? 8 : 0;
    const int kb  = (lane & 16) ? 8 : 0;
    const int bko = (lane & 16) ? 8 : 0;

    // Pre-gather Q A-fragments for both 32-wide K-steps over HD=64.
    Frag16 aq[2];
    {
        int am = wave * 16 + nl;
        #pragma unroll
        for (int ks = 0; ks < 2; ++ks) {
            aq[ks].q[0] = *(const v4u*)&Qs[am][ks * 32 + kb];
            aq[ks].q[1] = *(const v4u*)&Qs[am][ks * 32 + kb + 16];
        }
    }

    float m_r[8], l_r[8];
    v8f o[4];
    #pragma unroll
    for (int r = 0; r < 8; ++r) { m_r[r] = -__builtin_inff(); l_r[r] = 0.0f; }
    #pragma unroll
    for (int nt = 0; nt < 4; ++nt) o[nt] = vzero8();

    const int qw   = q0 + wave * 16;
    const int jend = (q0 + 127) >> 6;        // last 64-wide k tile this block needs

    for (int j = 0; j <= jend; ++j) {
        // stage K via async copy; V via load + LDS transpose
        #pragma unroll
        for (int it = 0; it < 2; ++it) {
            int i = tid + it * 256;
            int r = i >> 3, c8 = (i & 7) * 8;
            size_t g = base + (size_t)(j * 64 + r) * HHD + c8;
            async_copy_b128(ks_base + (unsigned)(r * LDK + c8) * 2u, &kws[g]);
            Pack8 t; t.q = *(const v4u*)&vws[g];
            #pragma unroll
            for (int jx = 0; jx < 8; ++jx) VsT[c8 + jx][r] = t.us[jx];
        }
        wait_async0();
        __syncthreads();

        if (j * 64 <= qw + 15) {             // wave-uniform: EXEC stays all-ones
            // S = Q @ K^T  (four 16-col subtiles, two K-steps over HD)
            v8f s[4];
            #pragma unroll
            for (int sub = 0; sub < 4; ++sub) s[sub] = vzero8();
            #pragma unroll
            for (int sub = 0; sub < 4; ++sub) {
                #pragma unroll
                for (int ks = 0; ks < 2; ++ks) {
                    Frag16 bkf;
                    bkf.q[0] = *(const v4u*)&Ks[sub * 16 + nl][ks * 32 + bko];
                    bkf.q[1] = *(const v4u*)&Ks[sub * 16 + nl][ks * 32 + bko + 16];
                    s[sub] = __builtin_amdgcn_wmma_f32_16x16x32_bf16(
                        false, aq[ks].v, false, bkf.v, (short)0, s[sub], false, false);
                }
            }

            // scale, causal mask, online softmax (rows span 16-lane halves)
            #pragma unroll
            for (int r = 0; r < 8; ++r) {
                int row = qw + hi8 + r;
                float sv[4];
                float lm = -__builtin_inff();
                #pragma unroll
                for (int sub = 0; sub < 4; ++sub) {
                    float t = s[sub][r] * 0.125f;            // 1/sqrt(64)
                    int c = j * 64 + sub * 16 + nl;
                    if (c > row) t = -__builtin_inff();
                    sv[sub] = t;
                    lm = fmaxf(lm, t);
                }
                #pragma unroll
                for (int ofs = 1; ofs < 16; ofs <<= 1)
                    lm = fmaxf(lm, __shfl_xor(lm, ofs));
                float nm  = fmaxf(m_r[r], lm);
                float nma = (nm == -__builtin_inff()) ? 0.0f : nm;  // avoid inf-inf
                float alpha = __expf(m_r[r] - nma);                 // -inf -> 0
                float ls = 0.0f;
                #pragma unroll
                for (int sub = 0; sub < 4; ++sub) {
                    float p = __expf(sv[sub] - nma);                // -inf -> 0
                    ls += p;
                    Ps[wave][hi8 + r][sub * 16 + nl] = f2bf(p);     // C-layout -> LDS
                }
                #pragma unroll
                for (int ofs = 1; ofs < 16; ofs <<= 1)
                    ls += __shfl_xor(ls, ofs);
                l_r[r] = l_r[r] * alpha + ls;
                m_r[r] = nm;
                #pragma unroll
                for (int nt = 0; nt < 4; ++nt) o[nt][r] *= alpha;
            }

            // O += P @ V   (P via per-wave LDS; same-wave DS ops are in order)
            #pragma unroll
            for (int ks = 0; ks < 2; ++ks) {
                Frag16 ap;
                ap.q[0] = *(const v4u*)&Ps[wave][nl][ks * 32 + kb];
                ap.q[1] = *(const v4u*)&Ps[wave][nl][ks * 32 + kb + 16];
                #pragma unroll
                for (int nt = 0; nt < 4; ++nt) {
                    Frag16 bvf;
                    bvf.q[0] = *(const v4u*)&VsT[nt * 16 + nl][ks * 32 + bko];
                    bvf.q[1] = *(const v4u*)&VsT[nt * 16 + nl][ks * 32 + bko + 16];
                    o[nt] = __builtin_amdgcn_wmma_f32_16x16x32_bf16(
                        false, ap.v, false, bvf.v, (short)0, o[nt], false, false);
                }
            }
        }
        __syncthreads();
    }

    #pragma unroll
    for (int nt = 0; nt < 4; ++nt) {
        int col = nt * 16 + nl;
        #pragma unroll
        for (int r = 0; r < 8; ++r) {
            int row = qw + hi8 + r;
            zws[base + (size_t)row * HHD + col] = f2bf(o[nt][r] / l_r[r]);
        }
    }
}

// ---------------------------------------------------------------------------
// Kernel 3: output projection  out[bs,m] = z @ W_O + b_O   (fp32 out)
// ---------------------------------------------------------------------------
__global__ __launch_bounds__(256) void out_proj_kernel(
    const unsigned short* __restrict__ zws,
    const float* __restrict__ Wo,
    const float* __restrict__ bo,
    float* __restrict__ out)
{
    __shared__ __align__(16) unsigned short As [128][LDK];
    __shared__ __align__(16) unsigned short BsT[64][LDK];

    const int tid  = threadIdx.x;
    const int wave = tid >> 5;
    const int lane = tid & 31;
    const int m0   = blockIdx.x * 128;
    const int n0   = blockIdx.y * 64;

    v8f acc[4];
    #pragma unroll
    for (int nt = 0; nt < 4; ++nt) acc[nt] = vzero8();

    const int nl  = lane & 15;
    const int am  = wave * 16 + nl;
    const int kb  = (lane & 16) ? 8 : 0;
    const int bko = (lane & 16) ? 8 : 0;

    for (int k0 = 0; k0 < HHD; k0 += 64) {
        // stage A: already bf16, straight copy (b128 in / b128 out)
        #pragma unroll
        for (int it = 0; it < 4; ++it) {
            int i = tid + it * 256;
            int r = i >> 3, c8 = (i & 7) * 8;
            *(v4u*)&As[r][c8] = *(const v4u*)&zws[(size_t)(m0 + r) * HHD + (k0 + c8)];
        }
        // stage B transposed: Wo[k][n] -> BsT[n][k]
        #pragma unroll
        for (int it = 0; it < 4; ++it) {
            int i = tid + it * 256;
            int r = i >> 4, c4 = (i & 15) * 4;
            v4f wv = *(const v4f*)&Wo[(size_t)(k0 + r) * Dc + (n0 + c4)];
            #pragma unroll
            for (int jx = 0; jx < 4; ++jx) BsT[c4 + jx][r] = f2bf(wv[jx]);
        }
        __syncthreads();

        #pragma unroll
        for (int ks = 0; ks < 2; ++ks) {
            Frag16 a;
            a.q[0] = *(const v4u*)&As[am][ks * 32 + kb];
            a.q[1] = *(const v4u*)&As[am][ks * 32 + kb + 16];
            #pragma unroll
            for (int nt = 0; nt < 4; ++nt) {
                Frag16 bfr;
                bfr.q[0] = *(const v4u*)&BsT[nt * 16 + nl][ks * 32 + bko];
                bfr.q[1] = *(const v4u*)&BsT[nt * 16 + nl][ks * 32 + bko + 16];
                acc[nt] = __builtin_amdgcn_wmma_f32_16x16x32_bf16(
                    false, a.v, false, bfr.v, (short)0, acc[nt], false, false);
            }
        }
        __syncthreads();
    }

    const int rbase = m0 + wave * 16 + ((lane & 16) ? 8 : 0);
    #pragma unroll
    for (int nt = 0; nt < 4; ++nt) {
        int col = n0 + nt * 16 + nl;
        float bb = bo[col];
        #pragma unroll
        for (int r = 0; r < 8; ++r)
            out[(size_t)(rbase + r) * Dc + col] = acc[nt][r] + bb;
    }
}

// ---------------------------------------------------------------------------
// Launch: qkv_proj -> flash attention -> out_proj.
// Workspace (bf16): q | k | v | z, each 4096x1024 = 8 MB -> 32 MB total.
// ---------------------------------------------------------------------------
extern "C" void kernel_launch(void* const* d_in, const int* in_sizes, int n_in,
                              void* d_out, int out_size, void* d_ws, size_t ws_size,
                              hipStream_t stream) {
    (void)in_sizes; (void)n_in; (void)out_size; (void)ws_size;

    const float* x  = (const float*)d_in[0];
    const float* Wq = (const float*)d_in[1];
    const float* Wk = (const float*)d_in[2];
    const float* Wv = (const float*)d_in[3];
    const float* Wo = (const float*)d_in[4];
    const float* bq = (const float*)d_in[5];
    const float* bk = (const float*)d_in[6];
    const float* bv = (const float*)d_in[7];
    const float* bo = (const float*)d_in[8];

    unsigned short* qws = (unsigned short*)d_ws;
    unsigned short* kws = qws + (size_t)ROWS * HHD;
    unsigned short* vws = kws + (size_t)ROWS * HHD;
    unsigned short* zws = vws + (size_t)ROWS * HHD;

    qkv_proj_kernel<<<dim3(ROWS / 128, Hc, 3), 256, 0, stream>>>(
        x, Wq, Wk, Wv, bq, bk, bv, qws, kws, vws);

    attn_kernel<<<dim3(Sc / 128, Hc, Bc), 256, 0, stream>>>(qws, kws, vws, zws);

    out_proj_kernel<<<dim3(ROWS / 128, Dc / 64), 256, 0, stream>>>(
        zws, Wo, bo, (float*)d_out);
}
